// GraphTransformerEncoder_39539468927051
// MI455X (gfx1250) — compile-verified, matrix operand order checked by reading, not run on hardware
//
#include <hip/hip_runtime.h>
#include <hip/hip_bf16.h>

#define D 128

typedef __attribute__((ext_vector_type(16))) __bf16 v16bf;
typedef __attribute__((ext_vector_type(8)))  float  v8f;
typedef int v4i __attribute__((vector_size(16)));   // matches builtin param type

// ---------------------------------------------------------- CDNA5 async copy
// 16B global -> LDS async copy (ASYNCcnt-tracked), per CDNA5 ISA.
__device__ __forceinline__ void async_load16(const void* g, void* l) {
#if __has_builtin(__builtin_amdgcn_global_load_async_to_lds_b128)
  __builtin_amdgcn_global_load_async_to_lds_b128((v4i*)g, (v4i*)l, 0, 0);
#else
  asm volatile("global_load_async_to_lds_b128 %0, %1, off"
               :: "v"((unsigned)(unsigned long long)l),
                  "v"((unsigned long long)g)
               : "memory");
#endif
}

__device__ __forceinline__ void wait_async0() {
#if __has_builtin(__builtin_amdgcn_s_wait_asynccnt)
  __builtin_amdgcn_s_wait_asynccnt(0);
#else
  asm volatile("s_wait_asynccnt 0x0" ::: "memory");
#endif
}

// ---------------------------------------------------------------- embeddings
__global__ void embed_kernel(const int* __restrict__ x,
                             const float* __restrict__ emb1,
                             const float* __restrict__ emb2,
                             float* __restrict__ oh, int N) {
  int idx = blockIdx.x * blockDim.x + threadIdx.x;
  if (idx >= N * D) return;
  int i = idx >> 7, d = idx & (D - 1);
  oh[idx] = emb1[x[2 * i] * D + d] + emb2[x[2 * i + 1] * D + d];
}

__global__ void gather_kernel(const float* __restrict__ oh,
                              const int* __restrict__ nidx,
                              float* __restrict__ h, int M) {
  long long idx = (long long)blockIdx.x * blockDim.x + threadIdx.x;
  if (idx >= (long long)M * D) return;
  int m = (int)(idx >> 7), d = (int)(idx & (D - 1));
  h[idx] = oh[(long long)nidx[m] * D + d];
}

// ------------------------------------------------------------- aggregation
__global__ void agg_init_kernel(const float* __restrict__ h,
                                const float* __restrict__ e1,
                                const float* __restrict__ e2,
                                float* __restrict__ agg, int M) {
  long long idx = (long long)blockIdx.x * blockDim.x + threadIdx.x;
  if (idx >= (long long)M * D) return;
  int d = (int)(idx & (D - 1));
  agg[idx] = h[idx] + e1[4 * D + d] + e2[d];
}

__global__ void agg_edge_kernel(const float* __restrict__ h,
                                const int* __restrict__ src,
                                const int* __restrict__ dst,
                                const int* __restrict__ attr,
                                const float* __restrict__ e1,
                                const float* __restrict__ e2,
                                float* __restrict__ agg, int E) {
  long long idx = (long long)blockIdx.x * blockDim.x + threadIdx.x;
  if (idx >= (long long)E * D) return;
  int e = (int)(idx >> 7), d = (int)(idx & (D - 1));
  int s = src[e], t = dst[e];
  float v = h[(long long)s * D + d] + e1[attr[2 * e] * D + d]
          + e2[attr[2 * e + 1] * D + d];
  atomicAdd(&agg[(long long)t * D + d], v);
}

__global__ void scatter_kernel(const float* __restrict__ h,
                               const int* __restrict__ ind,
                               float* __restrict__ xs, int M) {
  long long idx = (long long)blockIdx.x * blockDim.x + threadIdx.x;
  if (idx >= (long long)M * D) return;
  int m = (int)(idx >> 7), d = (int)(idx & (D - 1));
  atomicAdd(&xs[(long long)ind[m] * D + d], h[idx]);
}

__global__ void concat_kernel(const float* __restrict__ oh,
                              const float* __restrict__ xs,
                              float* __restrict__ cat, int N) {
  int idx = blockIdx.x * blockDim.x + threadIdx.x;
  if (idx >= N * 2 * D) return;
  int r = idx >> 8, c = idx & (2 * D - 1);
  cat[idx] = (c < D) ? oh[r * D + c] : xs[r * D + (c - D)];
}

// ------------------------------------------------------------ weight prep
__global__ void transpose_bf16_kernel(const float* __restrict__ W,
                                      __bf16* __restrict__ Wt,
                                      int K, int Nout) {
  int idx = blockIdx.x * blockDim.x + threadIdx.x;
  if (idx >= K * Nout) return;
  int n = idx / K, k = idx - n * K;
  Wt[idx] = (__bf16)W[(long long)k * Nout + n];
}

// ---------------------------------------------------------------- WMMA GEMM
// C[M x Nout] = act(A[M x K] @ Wt^T + bias); Wt is [Nout x K] bf16.
// KT = K/32 (compile-time); ABF16: A already bf16 -> async global->LDS copy;
// CBF16: write C as bf16 (feeding the next WMMA GEMM).
template <int KT, bool ABF16, bool CBF16>
__global__ void gemm_bias_act_kernel(const void* __restrict__ Ain,
                                     const __bf16* __restrict__ Wt,
                                     const float* __restrict__ bias,
                                     void* __restrict__ Cout,
                                     int M, int Nout, int relu) {
  constexpr int K   = KT * 32;
  constexpr int LDA = K + 8;               // pad 16B: breaks LDS bank conflicts
  extern __shared__ __bf16 sA[];           // 128 x LDA bf16
  const int blockRow = blockIdx.x * 128;
  const int tid = threadIdx.x;

  // ---- stage A tile into LDS
  if (ABF16) {
    const __bf16* A = (const __bf16*)Ain;
    if (blockRow + 128 <= M) {
      // pure byte copy: CDNA5 async global->LDS, 16B per lane per op
      constexpr int cpr = K >> 3;          // 16B chunks per row
      for (int i = tid; i < 128 * cpr; i += 256) {
        int r = i / cpr, c = (i - r * cpr) << 3;
        async_load16(A + (long long)(blockRow + r) * K + c, &sA[r * LDA + c]);
      }
      wait_async0();
    } else {                               // edge block: manual, zero-padded
      for (int i = tid; i < 128 * K; i += 256) {
        int r = i / K, c = i - r * K;
        int row = blockRow + r;
        sA[r * LDA + c] = (row < M) ? A[(long long)row * K + c] : (__bf16)0.0f;
      }
    }
  } else {
    const float* A = (const float*)Ain;
    for (int i = tid; i < 128 * K; i += 256) {
      int r = i / K, c = i - r * K;
      int row = blockRow + r;
      float v = (row < M) ? A[(long long)row * K + c] : 0.0f;
      sA[r * LDA + c] = (__bf16)v;
    }
  }
  __syncthreads();

  const int wave = tid >> 5;
  const int lane = tid & 31;
  const int r0   = wave << 4;              // wave's 16-row stripe
  const int mrow = lane & 15;              // A row / B col / C col within tile
  const int kh   = (lane >> 4) << 3;       // 0 or 8
  const int ntiles = Nout >> 4;

  // Preload ALL A fragments once: invariant across every n-tile.
  v16bf af[KT];
#pragma unroll
  for (int kt = 0; kt < KT; ++kt) {
    const __bf16* ap = &sA[(r0 + mrow) * LDA + (kt << 5) + kh];
#pragma unroll
    for (int i = 0; i < 8; ++i) { af[kt][i] = ap[i]; af[kt][i + 8] = ap[i + 16]; }
  }

  for (int nt = 0; nt < ntiles; ++nt) {
    const int ncol = (nt << 4) + mrow;
    const __bf16* wrow = Wt + (long long)ncol * K;
    if (nt + 1 < ntiles)
      __builtin_prefetch(wrow + 16LL * K, 0, 1);

    // Batch-load all B fragments for this n-tile (clause-able), then chain WMMAs.
    v16bf bf[KT];
#pragma unroll
    for (int kt = 0; kt < KT; ++kt) {
      const __bf16* bp = &wrow[(kt << 5) + kh];
#pragma unroll
      for (int i = 0; i < 8; ++i) { bf[kt][i] = bp[i]; bf[kt][i + 8] = bp[i + 16]; }
    }

    v8f acc = {};
#pragma unroll
    for (int kt = 0; kt < KT; ++kt)
      acc = __builtin_amdgcn_wmma_f32_16x16x32_bf16(
          false, af[kt], false, bf[kt], (short)0, acc, false, false);

    const float bv = bias[ncol];
    const int rbase = blockRow + r0 + ((lane >> 4) << 3);
#pragma unroll
    for (int g = 0; g < 8; ++g) {
      int row = rbase + g;
      if (row < M) {
        float v = acc[g] + bv;
        if (relu) v = fmaxf(v, 0.0f);
        if (CBF16) ((__bf16*)Cout)[(long long)row * Nout + ncol] = (__bf16)v;
        else       ((float*)Cout)[(long long)row * Nout + ncol] = v;
      }
    }
  }
}

// ------------------------------------------------------------- batch-norm
__global__ void colstats_partial_kernel(const float* __restrict__ X, int N, int C,
                                        float* __restrict__ sums,
                                        float* __restrict__ sumsq) {
  int c = threadIdx.x;                      // blockDim.x == C
  float s = 0.0f, ss = 0.0f;
  for (int r = blockIdx.x; r < N; r += gridDim.x) {
    float v = X[(long long)r * C + c];
    s += v; ss += v * v;
  }
  atomicAdd(&sums[c], s);
  atomicAdd(&sumsq[c], ss);
}

__global__ void colstats_finalize_kernel(const float* __restrict__ sums,
                                         const float* __restrict__ sumsq,
                                         int N, int C,
                                         float* __restrict__ mean,
                                         float* __restrict__ rstd) {
  int c = blockIdx.x * blockDim.x + threadIdx.x;
  if (c >= C) return;
  float m = sums[c] / (float)N;
  float v = sumsq[c] / (float)N - m * m;
  mean[c] = m;
  rstd[c] = rsqrtf(v + 1e-5f);
}

__global__ void bn_apply_kernel(const float* __restrict__ X,
                                const float* __restrict__ mean,
                                const float* __restrict__ rstd,
                                const float* __restrict__ gamma,
                                const float* __restrict__ beta,
                                void* __restrict__ out, int out_bf16,
                                long long total, int cmask) {
  long long idx = (long long)blockIdx.x * blockDim.x + threadIdx.x;
  if (idx >= total) return;
  int c = (int)(idx & cmask);
  float v = (X[idx] - mean[c]) * rstd[c] * gamma[c] + beta[c];
  if (out_bf16) ((__bf16*)out)[idx] = (__bf16)v;
  else          ((float*)out)[idx] = v;
}

// ------------------------------------------------------------------ driver
extern "C" void kernel_launch(void* const* d_in, const int* in_sizes, int n_in,
                              void* d_out, int out_size, void* d_ws, size_t ws_size,
                              hipStream_t stream) {
  const int*   x        = (const int*)d_in[0];
  const int*   sub_node = (const int*)d_in[3];
  const int*   sub_edge = (const int*)d_in[4];
  const int*   sub_attr = (const int*)d_in[5];
  const int*   sub_ind  = (const int*)d_in[6];
  const float* x_emb1   = (const float*)d_in[7];
  const float* x_emb2   = (const float*)d_in[8];
  const float* ee1      = (const float*)d_in[9];
  const float* ee2      = (const float*)d_in[10];
  const float* W1       = (const float*)d_in[11];
  const float* b1       = (const float*)d_in[12];
  const float* W2       = (const float*)d_in[13];
  const float* b2       = (const float*)d_in[14];
  const float* bn1g     = (const float*)d_in[15];
  const float* bn1b     = (const float*)d_in[16];
  const float* Wp       = (const float*)d_in[17];
  const float* bp       = (const float*)d_in[18];
  const float* bn2g     = (const float*)d_in[19];
  const float* bn2b     = (const float*)d_in[20];

  const int N = in_sizes[0] / 2;
  const int M = in_sizes[3];
  const int E = in_sizes[5] / 2;
  const int* src = sub_edge;
  const int* dst = sub_edge + E;

  // ----- workspace carve-out
  char* ws = (char*)d_ws;
  size_t off = 0;
  auto carve = [&](size_t bytes) -> void* {
    void* p = ws + off;
    off += (bytes + 255) & ~(size_t)255;
    return p;
  };
  float*  h    = (float*)carve((size_t)M * D * 4);
  float*  agg  = (float*)carve((size_t)M * D * 4);
  __bf16* hid  = (__bf16*)carve((size_t)M * 2 * D * 2);      // bf16 hidden
  float*  oh   = (float*)carve((size_t)N * D * 4);
  float*  xs   = (float*)carve((size_t)N * D * 4);
  float*  cat  = (float*)carve((size_t)N * 2 * D * 4);
  __bf16* catn = (__bf16*)carve((size_t)N * 2 * D * 2);      // bn1(cat) in bf16
  float*  proj = (float*)carve((size_t)N * D * 4);
  __bf16* wt1  = (__bf16*)carve((size_t)2 * 2 * D * D * 2);
  __bf16* wt2  = (__bf16*)carve((size_t)2 * D * 2 * D * 2);
  __bf16* wtp  = (__bf16*)carve((size_t)D * 2 * D * 2);
  float*  sums = (float*)carve(256 * 4);
  float*  sumsq= (float*)carve(256 * 4);
  float*  mean = (float*)carve(256 * 4);
  float*  rstd = (float*)carve(256 * 4);
  (void)ws_size; (void)n_in; (void)out_size;

  const int TB = 256;
  auto blocks = [&](long long n) { return (int)((n + TB - 1) / TB); };
  const int SH_K128 = 128 * (D + 8) * 2;       // 34,816 B
  const int SH_K256 = 128 * (2 * D + 8) * 2;   // 67,584 B (LDS is 320KB/WGP)

  // ----- weight prep (f32 -> transposed bf16)
  for (int l = 0; l < 2; ++l) {
    transpose_bf16_kernel<<<blocks(D * 2 * D), TB, 0, stream>>>(
        W1 + (size_t)l * D * 2 * D, wt1 + (size_t)l * 2 * D * D, D, 2 * D);
    transpose_bf16_kernel<<<blocks(2 * D * D), TB, 0, stream>>>(
        W2 + (size_t)l * 2 * D * D, wt2 + (size_t)l * D * 2 * D, 2 * D, D);
  }
  transpose_bf16_kernel<<<blocks(2 * D * D), TB, 0, stream>>>(Wp, wtp, 2 * D, D);

  // ----- node embeddings + subgraph gather
  embed_kernel<<<blocks((long long)N * D), TB, 0, stream>>>(x, x_emb1, x_emb2, oh, N);
  gather_kernel<<<blocks((long long)M * D), TB, 0, stream>>>(oh, sub_node, h, M);

  // ----- 2 GIN layers
  for (int l = 0; l < 2; ++l) {
    const float* e1 = ee1 + (size_t)l * 6 * D;
    const float* e2 = ee2 + (size_t)l * 3 * D;
    agg_init_kernel<<<blocks((long long)M * D), TB, 0, stream>>>(h, e1, e2, agg, M);
    agg_edge_kernel<<<blocks((long long)E * D), TB, 0, stream>>>(
        h, src, dst, sub_attr, e1, e2, agg, E);

    // hidden = relu(agg @ W1 + b1): K=128, A f32 (converted), C bf16
    gemm_bias_act_kernel<4, false, true><<<(M + 127) / 128, TB, SH_K128, stream>>>(
        agg, wt1 + (size_t)l * 2 * D * D, b1 + (size_t)l * 2 * D, hid,
        M, 2 * D, 1);
    // h = relu(hidden @ W2 + b2): K=256, A bf16 (async-to-LDS), C f32
    gemm_bias_act_kernel<8, true, false><<<(M + 127) / 128, TB, SH_K256, stream>>>(
        hid, wt2 + (size_t)l * D * 2 * D, b2 + (size_t)l * D, h,
        M, D, 1);
  }

  // ----- scatter back to centers, concat
  (void)hipMemsetAsync(xs, 0, (size_t)N * D * 4, stream);
  scatter_kernel<<<blocks((long long)M * D), TB, 0, stream>>>(h, sub_ind, xs, M);
  concat_kernel<<<blocks((long long)N * 2 * D), TB, 0, stream>>>(oh, xs, cat, N);

  // ----- bn1 over [N x 256] -> bf16 activations for projection GEMM
  (void)hipMemsetAsync(sums, 0, 256 * 4, stream);
  (void)hipMemsetAsync(sumsq, 0, 256 * 4, stream);
  colstats_partial_kernel<<<256, 2 * D, 0, stream>>>(cat, N, 2 * D, sums, sumsq);
  colstats_finalize_kernel<<<1, 2 * D, 0, stream>>>(sums, sumsq, N, 2 * D, mean, rstd);
  bn_apply_kernel<<<blocks((long long)N * 2 * D), TB, 0, stream>>>(
      cat, mean, rstd, bn1g, bn1b, catn, 1, (long long)N * 2 * D, 2 * D - 1);

  // ----- projection: proj = bn1(cat) @ Wp + bp   (K=256, A bf16 async)
  gemm_bias_act_kernel<8, true, false><<<(N + 127) / 128, TB, SH_K256, stream>>>(
      catn, wtp, bp, proj, N, D, 0);

  // ----- bn2 over [N x 128] -> d_out (f32)
  (void)hipMemsetAsync(sums, 0, 128 * 4, stream);
  (void)hipMemsetAsync(sumsq, 0, 128 * 4, stream);
  colstats_partial_kernel<<<256, D, 0, stream>>>(proj, N, D, sums, sumsq);
  colstats_finalize_kernel<<<1, D, 0, stream>>>(sums, sumsq, N, D, mean, rstd);
  bn_apply_kernel<<<blocks((long long)N * D), TB, 0, stream>>>(
      proj, mean, rstd, bn2g, bn2b, d_out, 0, (long long)N * D, D - 1);
}
